// ResidualVectorQuantizer_23167053595189
// MI455X (gfx1250) — matrix operand order repaired and checked by reading, not compiled, not run on hardware
//
#include <hip/hip_runtime.h>

#define DD 512
#define QQ 8
#define KK 2048
#define BB 8
#define LL 2048
#define PADW 520  // ushort stride per code row in LDS (pad 8 -> 16B aligned, bank-spread)

typedef __bf16 v16bf __attribute__((ext_vector_type(16)));
typedef float  v8f   __attribute__((ext_vector_type(8)));
typedef int    v4i_t __attribute__((ext_vector_type(4)));

#if __has_builtin(__builtin_amdgcn_global_load_async_to_lds_b128)
#define HAVE_ASYNC_LDS 1
#endif

__device__ __forceinline__ void wait_async0(){
#if defined(HAVE_ASYNC_LDS)
#if __has_builtin(__builtin_amdgcn_s_wait_asynccnt)
  __builtin_amdgcn_s_wait_asynccnt(0);
#else
  asm volatile("s_wait_asynccnt 0x0" ::: "memory");
#endif
#endif
}

// Stage 64B (4 x b128) from global to LDS for this thread.
__device__ __forceinline__ void stage64(const unsigned short* gsrc, unsigned short* ldst){
#if defined(HAVE_ASYNC_LDS)
  #pragma unroll
  for (int i = 0; i < 4; ++i){
    __builtin_amdgcn_global_load_async_to_lds_b128(
        (v4i_t*)(gsrc + i * 8), (v4i_t*)(ldst + i * 8), 0, 0);
  }
#else
  const uint4* s4 = (const uint4*)gsrc;
  uint4* d4 = (uint4*)ldst;
  uint4 a0 = s4[0], a1 = s4[1], a2 = s4[2], a3 = s4[3];
  d4[0] = a0; d4[1] = a1; d4[2] = a2; d4[3] = a3;
#endif
}

__device__ __forceinline__ unsigned short f2bf(float f){
  unsigned int u = __float_as_uint(f);
  u += 0x7FFFu + ((u >> 16) & 1u);   // round-to-nearest-even truncation to bf16
  return (unsigned short)(u >> 16);
}

// One wave per code row: fp32 -> bf16 codebook + ||c||^2.
__global__ __launch_bounds__(256) void rvq_prep(const float* __restrict__ cb,
                                                unsigned short* __restrict__ cb_bf,
                                                float* __restrict__ cb2){
  int gwave = (blockIdx.x * blockDim.x + threadIdx.x) >> 5;
  int lane  = threadIdx.x & 31;
  if (gwave >= QQ * KK) return;
  const float* row = cb + (size_t)gwave * DD;
  unsigned short* orow = cb_bf + (size_t)gwave * DD;
  float s = 0.f;
  #pragma unroll
  for (int i = 0; i < DD/32; ++i){
    float v = row[lane + 32*i];
    orow[lane + 32*i] = f2bf(v);
    s += v * v;
  }
  #pragma unroll
  for (int m = 1; m < 32; m <<= 1) s += __shfl_xor(s, m, 32);
  if (lane == 0) cb2[gwave] = s;
}

__global__ __launch_bounds__(256) void rvq_main(
    const float* __restrict__ x,              // [B*L, D] fp32
    const float* __restrict__ cb,             // [Q, K, D] fp32
    const unsigned short* __restrict__ cb_bf, // [Q, K, D] bf16 bits
    const float* __restrict__ cb2,            // [Q, K]
    float* __restrict__ resid,                // [B*L, D] scratch
    float* __restrict__ out_idx,              // [B, Q, L] (as float)
    float* __restrict__ out_quant)            // [B*L, D]
{
  __shared__ unsigned short sB[2][16 * PADW];   // double-buffered 16-code bf16 tile

  const int t      = threadIdx.x;
  const int lane   = t & 31;
  const int wv     = t >> 5;                    // wave in block, 0..7
  const int n_lane = lane & 15;
  const bool hi    = lane >= 16;
  const int tile   = blockIdx.x * 8 + wv;       // 0..1023 (16-row M tiles)
  const int row0   = tile * 16;

  // cooperative tile-staging role: 256 threads cover 16 rows x 16 chunks of 64B
  const int lr = t >> 4;                        // code row within tile
  const int lc = (t & 15) * 32;                 // ushort column offset

  v16bf A[16];
  float best[8];
  int   bidx[8];

  for (int q = 0; q < QQ; ++q){
    const unsigned short* cbq  = cb_bf + (size_t)q * KK * DD;
    const float*          cb2q = cb2 + q * KK;
    const float*          rsrc = (q == 0) ? x : resid;

    // ---- build bf16 A fragments (16x32 per fragment) from fp32 residual ----
    {
      const float* rowp = rsrc + (size_t)(row0 + n_lane) * DD;
      #pragma unroll
      for (int kc = 0; kc < 16; ++kc){
        const int dbase = kc * 32 + (hi ? 16 : 0);
        union { v16bf v; unsigned short s[16]; } af;
        #pragma unroll
        for (int e = 0; e < 8; ++e) af.s[e]     = f2bf(rowp[dbase + e]);
        #pragma unroll
        for (int e = 0; e < 8; ++e) af.s[8 + e] = f2bf(rowp[dbase + 8 + e]);
        A[kc] = af.v;
      }
    }
    #pragma unroll
    for (int v = 0; v < 8; ++v){ best[v] = 3.4e38f; bidx[v] = 0; }

    // ---- prologue: stage code tile 0 into buffer 0 (async if available) ----
    stage64(cbq + (size_t)lr * DD + lc, &sB[0][lr * PADW + lc]);
    wait_async0();
    __syncthreads();

    for (int nt = 0; nt < KK/16; ++nt){
      // async-prefetch next 16-code tile into the other buffer (overlaps WMMAs)
      if (nt + 1 < KK/16){
        stage64(cbq + (size_t)((nt + 1) * 16 + lr) * DD + lc,
                &sB[(nt + 1) & 1][lr * PADW + lc]);
      }
      // ---- 16 WMMAs over D=512; B fragments double-buffered in registers ----
      const unsigned short* bufp = sB[nt & 1] + n_lane * PADW + (hi ? 16 : 0);
      union BF { v16bf v; uint4 u[2]; };
      BF bfr[2];
      {
        const uint4* p = (const uint4*)bufp;
        bfr[0].u[0] = p[0]; bfr[0].u[1] = p[1];
      }
      v8f acc = {0.f,0.f,0.f,0.f,0.f,0.f,0.f,0.f};
      #pragma unroll
      for (int kc = 0; kc < 16; ++kc){
        if (kc + 1 < 16){
          const uint4* p = (const uint4*)(bufp + (kc + 1) * 32);
          bfr[(kc + 1) & 1].u[0] = p[0];
          bfr[(kc + 1) & 1].u[1] = p[1];
        }
        acc = __builtin_amdgcn_wmma_f32_16x16x32_bf16(false, A[kc], false, bfr[kc & 1].v,
                                                      (short)0, acc, false, false);
      }
      // dist ~ ||c||^2 - 2 r.c  (||r||^2 constant per row: irrelevant for argmin)
      const float c2   = cb2q[nt * 16 + n_lane];
      const int   code = nt * 16 + n_lane;
      #pragma unroll
      for (int v = 0; v < 8; ++v){
        float d2 = __builtin_fmaf(-2.0f, acc[v], c2);
        if (d2 < best[v]){ best[v] = d2; bidx[v] = code; }
      }
      wait_async0();        // prefetched tile has landed in LDS
      __syncthreads();
    }

    // ---- argmin across the 16 lanes of each half (C layout: lane half = M group) ----
    #pragma unroll
    for (int v = 0; v < 8; ++v){
      float bv = best[v]; int bi = bidx[v];
      #pragma unroll
      for (int m = 1; m < 16; m <<= 1){
        float ov = __shfl_xor(bv, m, 32);
        int   oi = __shfl_xor(bi, m, 32);
        if (ov < bv || (ov == bv && oi < bi)){ bv = ov; bi = oi; }  // first-index tie-break
      }
      best[v] = bv; bidx[v] = bi;
    }

    // ---- write indices: rows 0..7 live in lanes 0-15 (VGPR v), 8..15 in lanes 16-31 ----
    if (n_lane == 0){
      const int mbase = hi ? 8 : 0;
      #pragma unroll
      for (int v = 0; v < 8; ++v){
        int row = row0 + mbase + v;
        int b = row >> 11, lp = row & (LL - 1);
        out_idx[((size_t)b * QQ + q) * LL + lp] = (float)bidx[v];
      }
    }

    // ---- fp32 residual update + quantized accumulation (per-wave independent) ----
    #pragma unroll
    for (int m = 0; m < 16; ++m){
      const int idx = __shfl(bidx[m & 7], (m < 8) ? 0 : 16, 32);
      const int row = row0 + m;
      const float* crow = cb + ((size_t)q * KK + idx) * DD;
      const float* rrow = rsrc + (size_t)row * DD;
      float* rout = resid     + (size_t)row * DD;
      float* qout = out_quant + (size_t)row * DD;
      for (int c = lane; c < DD; c += 32){
        float cv = crow[c];
        rout[c] = rrow[c] - cv;
        if (q == 0) qout[c] = cv; else qout[c] += cv;
      }
    }
  }
}

extern "C" void kernel_launch(void* const* d_in, const int* in_sizes, int n_in,
                              void* d_out, int out_size, void* d_ws, size_t ws_size,
                              hipStream_t stream){
  const float* x  = (const float*)d_in[0];   // [B, L, D]
  const float* cb = (const float*)d_in[1];   // [Q, K, D]

  char* ws = (char*)d_ws;
  unsigned short* cb_bf = (unsigned short*)ws;                          // 16 MiB
  float* cb2   = (float*)(ws + (size_t)QQ * KK * DD * 2);               // 64 KiB
  float* resid = (float*)(ws + (size_t)QQ * KK * DD * 2 + (size_t)QQ * KK * 4); // 32 MiB

  float* out_idx   = (float*)d_out;                  // B*Q*L floats
  float* out_quant = out_idx + (size_t)BB * QQ * LL; // B*L*D floats

  rvq_prep<<<dim3((QQ * KK) / 8), dim3(256), 0, stream>>>(cb, cb_bf, cb2);
  rvq_main<<<dim3(128), dim3(256), 0, stream>>>(x, cb, cb_bf, cb2, resid,
                                                out_idx, out_quant);
}
